// DuelingDistributionalNetwork_24103356465503
// MI455X (gfx1250) — matrix operand order repaired and checked by reading, not compiled, not run on hardware
//
#include <hip/hip_runtime.h>
#include <cstdint>
#include <cstddef>

typedef _Float16 v16h __attribute__((ext_vector_type(16)));
typedef _Float16 v8h  __attribute__((ext_vector_type(8)));
typedef float    v8f  __attribute__((ext_vector_type(8)));

#define IN_DIM   1024
#define HID      256
#define OUT_DIM  1000
#define NUM_BINS 51
#define BATCH    2048
#define N_LEGAL  200

#define A_GROUPS 63                    // ceil(1000/16)
#define A_PAD    (A_GROUPS * 16)       // 1008 padded actions
#define NCOL_PAD (A_PAD * NUM_BINS)    // 51408 padded columns
#define GRP_COLS (16 * NUM_BINS)       // 816 columns per action-group
#define GRP_TILES NUM_BINS             // 51 WMMA n-tiles per group

// ---------------------------------------------------------------------------
// A-fragment loader: 16-bit A 16x32 layout (ISA 7.12.2). Lane L holds row
// m = L&15; halves 0..7 = K[8*half .. 8*half+7], halves 8..15 = K[16+8*half ..].
// Two contiguous 16-byte loads from a row-major f16 matrix.
// ---------------------------------------------------------------------------
__device__ __forceinline__ v16h load_a_frag(const _Float16* __restrict__ arow,
                                            int kt, int half) {
  const v8h lo = *(const v8h*)(arow + kt * 32 + 8 * half);
  const v8h hi = *(const v8h*)(arow + kt * 32 + 16 + 8 * half);
  return __builtin_shufflevector(lo, hi, 0, 1, 2, 3, 4, 5, 6, 7,
                                         8, 9, 10, 11, 12, 13, 14, 15);
}

// ---------------------------------------------------------------------------
// Prep kernels (run every launch; deterministic)
// ---------------------------------------------------------------------------
__global__ void k_f32_to_f16(const float* __restrict__ src,
                             _Float16* __restrict__ dst, int n) {
  int i = blockIdx.x * 256 + threadIdx.x;
  if (i < n) dst[i] = (_Float16)src[i];
}

// Pack W[K,N] (f32 row-major) into f16 B-fragment order:
// flat = ((nt*KT + kt)*32 + lane)*16 + e ; lane&15 = n-in-tile, h = lane>>4,
// K = 8h+e for e<8, 16+8h+(e-8) for e>=8. Zero-fills columns >= N (action padding).
__global__ void k_pack_b(const float* __restrict__ W, _Float16* __restrict__ out,
                         int K, int N, int Npad) {
  const int KT = K >> 5;
  const long total = (long)KT * (Npad >> 4) * 512;
  long i = (long)blockIdx.x * 256 + threadIdx.x;
  if (i >= total) return;
  int e    = (int)(i & 15);
  int lane = (int)((i >> 4) & 31);
  long rest = i >> 9;
  int kt = (int)(rest % KT);
  int nt = (int)(rest / KT);
  int k = kt * 32 + 8 * (lane >> 4) + e + 8 * (e >> 3);
  int n = nt * 16 + (lane & 15);
  float v = (n < N) ? W[(size_t)k * N + n] : 0.0f;
  out[i] = (_Float16)v;
}

__global__ void k_pad_bias(const float* __restrict__ b, float* __restrict__ out,
                           int N, int Npad) {
  int i = blockIdx.x * 256 + threadIdx.x;
  if (i < Npad) out[i] = (i < N) ? b[i] : 0.0f;
}

// mean[k][bin] = mean over 1000 actions of w_ao[k, a*51+bin]; row 256 = bias mean.
__global__ void k_mean_wao(const float* __restrict__ w_ao,
                           const float* __restrict__ b_ao,
                           float* __restrict__ mean) {
  int i = blockIdx.x * 256 + threadIdx.x;
  if (i >= (HID + 1) * NUM_BINS) return;
  int row = i / NUM_BINS, bin = i % NUM_BINS;
  const float* src = (row < HID) ? (w_ao + (size_t)row * (OUT_DIM * NUM_BINS) + bin)
                                 : (b_ao + bin);
  float s = 0.0f;
  for (int a = 0; a < OUT_DIM; ++a) s += src[(size_t)a * NUM_BINS];
  mean[i] = s * (1.0f / OUT_DIM);
}

// shift[b,bin] = (val_h@w_vo + b_vo) - (adv_h@w_ao_mean + b_ao_mean)
__global__ void k_shift(const _Float16* __restrict__ vh,
                        const _Float16* __restrict__ ah,
                        const float* __restrict__ w_vo,
                        const float* __restrict__ b_vo,
                        const float* __restrict__ mean,
                        float* __restrict__ shift) {
  int i = blockIdx.x * 256 + threadIdx.x;
  if (i >= BATCH * NUM_BINS) return;
  int b = i / NUM_BINS, bin = i % NUM_BINS;
  float s = b_vo[bin] - mean[HID * NUM_BINS + bin];
  const _Float16* vrow = vh + (size_t)b * HID;
  const _Float16* arow = ah + (size_t)b * HID;
  for (int k = 0; k < HID; ++k)
    s += (float)vrow[k] * w_vo[k * NUM_BINS + bin]
       - (float)arow[k] * mean[k * NUM_BINS + bin];
  shift[i] = s;
}

// ---------------------------------------------------------------------------
// Generic WMMA GEMM: C_f16[M,N] = relu(A_f16[M,K] @ Bpacked + bias).
// Block = 256 threads = 8 waves; wave w handles n-tile blockIdx.x*8+w,
// blockIdx.y = m-tile. N must be a multiple of 128, K of 32.
// ---------------------------------------------------------------------------
__global__ void __launch_bounds__(256)
k_gemm_relu_f16(const _Float16* __restrict__ A, const _Float16* __restrict__ Bp,
                const float* __restrict__ bias, _Float16* __restrict__ C,
                int M, int N, int K) {
  const int lane = threadIdx.x & 31;
  const int wave = threadIdx.x >> 5;
  const int half = lane >> 4;
  const int ncol = lane & 15;
  const int ntile = blockIdx.x * 8 + wave;
  const int mtile = blockIdx.y;
  const int KT = K >> 5;
  const int m = mtile * 16 + ncol;               // A-fragment row
  const _Float16* arow = A + (size_t)m * K;
  const _Float16* bptr = Bp + (size_t)ntile * KT * 512 + lane * 16;
  v8f acc = {};
  for (int kt = 0; kt < KT; ++kt) {
    v16h a = load_a_frag(arow, kt, half);
    v16h b = *(const v16h*)(bptr + (size_t)kt * 512);
    acc = __builtin_amdgcn_wmma_f32_16x16x32_f16(false, a, false, b,
                                                 (short)0, acc, false, false);
  }
  const int col = ntile * 16 + ncol;
  const float bs = bias[col];
#pragma unroll
  for (int v = 0; v < 8; ++v) {
    int row = mtile * 16 + 8 * half + v;
    float x = acc[v] + bs;
    x = x > 0.0f ? x : 0.0f;
    C[(size_t)row * N + col] = (_Float16)x;
  }
}

// ---------------------------------------------------------------------------
// Fused big kernel: adv tile GEMM (WMMA) + bias + shift -> LDS ->
// per-(row,action) 51-bin softmax, clamp(1e-5), expected value -> q_vals.
// Grid: (63 action-groups, 128 batch-tiles), block 256 = 8 waves.
// ---------------------------------------------------------------------------
__global__ void __launch_bounds__(256)
k_big(const _Float16* __restrict__ Ah, const _Float16* __restrict__ Bp,
      const float* __restrict__ biasPad, const float* __restrict__ shift,
      const float* __restrict__ v_range, float* __restrict__ qvals) {
  __shared__ float ldsQ[16 * GRP_COLS];       // 52224 B
  __shared__ float ldsShift[16 * NUM_BINS];   //  3264 B
  __shared__ float ldsBias[GRP_COLS];         //  3264 B
  __shared__ float ldsVr[NUM_BINS];
  const int tid = threadIdx.x;
  const int bx = blockIdx.x;                  // action group (16 actions)
  const int mtile = blockIdx.y;               // 16 batch rows

  // phase 0: stage bias / shift / v_range
  for (int i = tid; i < GRP_COLS; i += 256) ldsBias[i] = biasPad[bx * GRP_COLS + i];
  for (int i = tid; i < 16 * NUM_BINS; i += 256) {
    int r = i / NUM_BINS, b = i % NUM_BINS;
    ldsShift[i] = shift[(size_t)(mtile * 16 + r) * NUM_BINS + b];
  }
  if (tid < NUM_BINS) ldsVr[tid] = v_range[tid];
  __syncthreads();

  // phase 1: WMMA over 51 n-tiles, round-robin across 8 waves (wave-uniform guards)
  const int lane = tid & 31, wave = tid >> 5;
  const int half = lane >> 4, ncol = lane & 15;
  const int m = mtile * 16 + ncol;
  const _Float16* arow = Ah + (size_t)m * HID;
  v8f acc[7];
#pragma unroll
  for (int t = 0; t < 7; ++t) acc[t] = (v8f){};
#pragma unroll
  for (int kt = 0; kt < HID / 32; ++kt) {
    v16h a = load_a_frag(arow, kt, half);
#pragma unroll
    for (int t = 0; t < 7; ++t) {
      int nt = wave + 8 * t;
      if (nt < GRP_TILES) {
        size_t boff = (((size_t)(bx * GRP_TILES + nt)) * (HID / 32) + kt) * 512
                    + (size_t)lane * 16;
        v16h b = *(const v16h*)(Bp + boff);
        acc[t] = __builtin_amdgcn_wmma_f32_16x16x32_f16(false, a, false, b,
                                                        (short)0, acc[t],
                                                        false, false);
      }
    }
  }
#pragma unroll
  for (int t = 0; t < 7; ++t) {
    int nt = wave + 8 * t;
    if (nt < GRP_TILES) {
      int col = nt * 16 + ncol;
      int bin = col % NUM_BINS;
      float bs = ldsBias[col];
#pragma unroll
      for (int v = 0; v < 8; ++v) {
        int row = 8 * half + v;
        ldsQ[row * GRP_COLS + col] = acc[t][v] + bs + ldsShift[row * NUM_BINS + bin];
      }
    }
  }
  __syncthreads();

  // phase 2: one thread per (row, action)
  const int r = tid >> 4;
  const int aloc = tid & 15;
  const float* q = &ldsQ[r * GRP_COLS + aloc * NUM_BINS];
  float mx = q[0];
  for (int i = 1; i < NUM_BINS; ++i) mx = fmaxf(mx, q[i]);
  float s = 0.0f;
  for (int i = 0; i < NUM_BINS; ++i) s += __expf(q[i] - mx);
  float inv = 1.0f / s;
  float qv = 0.0f;
  for (int i = 0; i < NUM_BINS; ++i) {
    float p = fmaxf(__expf(q[i] - mx) * inv, 1e-5f);
    qv += p * ldsVr[i];
  }
  int ag = bx * 16 + aloc;
  if (ag < OUT_DIM)
    qvals[(size_t)(mtile * 16 + r) * OUT_DIM + ag] = qv;
}

// ---------------------------------------------------------------------------
// Legal-move masking
// ---------------------------------------------------------------------------
__global__ void k_fill(float* __restrict__ out, int n, float val) {
  int i = blockIdx.x * 256 + threadIdx.x;
  if (i < n) out[i] = val;
}

__global__ void k_scatter(const int* __restrict__ moves,
                          const float* __restrict__ qvals,
                          float* __restrict__ out) {
  int i = blockIdx.x * 256 + threadIdx.x;
  if (i >= BATCH * N_LEGAL) return;
  int b = i / N_LEGAL;
  int a = moves[i];
  float v = qvals[(size_t)b * OUT_DIM + a];
  out[(size_t)b * OUT_DIM + a] = (v == 0.0f) ? -__builtin_inff() : v;
}

// ---------------------------------------------------------------------------
extern "C" void kernel_launch(void* const* d_in, const int* in_sizes, int n_in,
                              void* d_out, int out_size, void* d_ws, size_t ws_size,
                              hipStream_t stream) {
  (void)in_sizes; (void)n_in; (void)out_size; (void)ws_size;
  const float* x     = (const float*)d_in[0];
  const float* w_in  = (const float*)d_in[1];
  const float* b_in  = (const float*)d_in[2];
  const float* w_ah  = (const float*)d_in[3];
  const float* b_ah  = (const float*)d_in[4];
  const float* w_ao  = (const float*)d_in[5];
  const float* b_ao  = (const float*)d_in[6];
  const float* w_vh  = (const float*)d_in[7];
  const float* b_vh  = (const float*)d_in[8];
  const float* w_vo  = (const float*)d_in[9];
  const float* b_vo  = (const float*)d_in[10];
  const float* v_rng = (const float*)d_in[11];
  const int*   moves = (const int*)d_in[12];
  float* out = (float*)d_out;

  char* wsp = (char*)d_ws;
  auto alloc = [&](size_t bytes) {
    char* p = wsp;
    wsp += (bytes + 255) & ~(size_t)255;
    return p;
  };
  _Float16* x16  = (_Float16*)alloc((size_t)BATCH * IN_DIM * 2);
  _Float16* pWin = (_Float16*)alloc((size_t)IN_DIM * HID * 2);
  _Float16* pWah = (_Float16*)alloc((size_t)HID * HID * 2);
  _Float16* pWvh = (_Float16*)alloc((size_t)HID * HID * 2);
  _Float16* pWao = (_Float16*)alloc((size_t)HID * NCOL_PAD * 2);   // 26.3 MB, fits L2
  _Float16* h16  = (_Float16*)alloc((size_t)BATCH * HID * 2);
  _Float16* ah16 = (_Float16*)alloc((size_t)BATCH * HID * 2);
  _Float16* vh16 = (_Float16*)alloc((size_t)BATCH * HID * 2);
  float* meanbuf = (float*)alloc((size_t)(HID + 1) * NUM_BINS * 4);
  float* biasPad = (float*)alloc((size_t)NCOL_PAD * 4);
  float* shift   = (float*)alloc((size_t)BATCH * NUM_BINS * 4);
  float* qvals   = (float*)alloc((size_t)BATCH * OUT_DIM * 4);

  auto cdiv = [](long a, long b) { return (unsigned)((a + b - 1) / b); };

  // prep
  k_f32_to_f16<<<cdiv((long)BATCH * IN_DIM, 256), 256, 0, stream>>>(x, x16, BATCH * IN_DIM);
  k_pack_b<<<cdiv((long)(IN_DIM / 32) * (HID / 16) * 512, 256), 256, 0, stream>>>(
      w_in, pWin, IN_DIM, HID, HID);
  k_pack_b<<<cdiv((long)(HID / 32) * (HID / 16) * 512, 256), 256, 0, stream>>>(
      w_ah, pWah, HID, HID, HID);
  k_pack_b<<<cdiv((long)(HID / 32) * (HID / 16) * 512, 256), 256, 0, stream>>>(
      w_vh, pWvh, HID, HID, HID);
  k_pack_b<<<cdiv((long)(HID / 32) * (NCOL_PAD / 16) * 512, 256), 256, 0, stream>>>(
      w_ao, pWao, HID, OUT_DIM * NUM_BINS, NCOL_PAD);
  k_pad_bias<<<cdiv(NCOL_PAD, 256), 256, 0, stream>>>(b_ao, biasPad,
                                                      OUT_DIM * NUM_BINS, NCOL_PAD);
  k_mean_wao<<<cdiv((long)(HID + 1) * NUM_BINS, 256), 256, 0, stream>>>(w_ao, b_ao, meanbuf);

  // trunk + branch hiddens (f16 WMMA GEMMs)
  k_gemm_relu_f16<<<dim3(HID / 128, BATCH / 16), 256, 0, stream>>>(
      x16, pWin, b_in, h16, BATCH, HID, IN_DIM);
  k_gemm_relu_f16<<<dim3(HID / 128, BATCH / 16), 256, 0, stream>>>(
      h16, pWah, b_ah, ah16, BATCH, HID, HID);
  k_gemm_relu_f16<<<dim3(HID / 128, BATCH / 16), 256, 0, stream>>>(
      h16, pWvh, b_vh, vh16, BATCH, HID, HID);

  // per-(batch,bin) shift = val - adv_mean  (mean folded into weights)
  k_shift<<<cdiv((long)BATCH * NUM_BINS, 256), 256, 0, stream>>>(
      vh16, ah16, w_vo, b_vo, meanbuf, shift);

  // fused adv GEMM + softmax + expected value
  k_big<<<dim3(A_GROUPS, BATCH / 16), 256, 0, stream>>>(
      ah16, pWao, biasPad, shift, v_rng, qvals);

  // legal-move mask
  k_fill<<<cdiv((long)BATCH * OUT_DIM, 256), 256, 0, stream>>>(
      out, BATCH * OUT_DIM, -__builtin_inff());
  k_scatter<<<cdiv((long)BATCH * N_LEGAL, 256), 256, 0, stream>>>(moves, qvals, out);
}
